// Attention_18116172054662
// MI455X (gfx1250) — matrix operand order repaired and checked by reading, not compiled, not run on hardware
//
#include <hip/hip_runtime.h>
#include <hip/hip_bf16.h>
#include <math.h>

typedef __bf16 bf16;
typedef __attribute__((ext_vector_type(16))) __bf16 v16bf;
typedef __attribute__((ext_vector_type(8)))  __bf16 v8bf;
typedef __attribute__((ext_vector_type(4)))  __bf16 v4bf;
typedef __attribute__((ext_vector_type(8)))  float  v8f;
typedef __attribute__((ext_vector_type(4)))  float  v4f;

constexpr int Bn = 4;     // batch
constexpr int S  = 2048;  // sequence
constexpr int D  = 1024;  // feature dim
constexpr int O  = 1024;  // output dim

// ---------------------------------------------------------------------------
// WMMA fragment helpers (layouts per CDNA5 ISA 7.12.2, wave32)
// A 16x32 bf16: lane = (m & 15) + 16*khalf; lane holds K = khalf*8+{0..7} and
//               16+khalf*8+{0..7}  -> two 16-byte chunks at k and k+16.
// B 32x16 bf16 loaded from an N-major (transposed) buffer BT[n][k]: lane
//   (n = lane&15, khalf = lane>>4) holds K = khalf*8+{0..7} and 16+khalf*8+{0..7}, same as A.
// C/D f32 16x16: vgpr r, lanes 0-15 -> M=r, lanes 16-31 -> M=8+r; N = lane&15.
// ---------------------------------------------------------------------------

__device__ inline v16bf load_a_bf16(const bf16* __restrict__ A, int lda,
                                    int row, int kh, int k0) {
  const bf16* q = A + (size_t)row * lda + k0 + kh * 8;
  v8bf c0 = *(const v8bf*)q;
  v8bf c1 = *(const v8bf*)(q + 16);
  v16bf r;
#pragma unroll
  for (int i = 0; i < 8; i++) { r[i] = c0[i]; r[8 + i] = c1[i]; }
  return r;
}

__device__ inline v16bf load_a_f32(const float* __restrict__ A, int lda,
                                   int row, int kh, int k0) {
  const float* q = A + (size_t)row * lda + k0 + kh * 8;
  v8f c0 = *(const v8f*)q;
  v8f c1 = *(const v8f*)(q + 16);
  v16bf r;
#pragma unroll
  for (int i = 0; i < 8; i++) { r[i] = (bf16)c0[i]; r[8 + i] = (bf16)c1[i]; }
  return r;
}

__device__ inline v16bf load_bt_bf16(const bf16* __restrict__ BT, int ldbt,
                                     int col, int kh, int k0) {
  const bf16* q = BT + (size_t)col * ldbt + k0 + kh * 8;
  v8bf c0 = *(const v8bf*)q;
  v8bf c1 = *(const v8bf*)(q + 16);
  v16bf r;
#pragma unroll
  for (int i = 0; i < 8; i++) { r[i] = c0[i]; r[8 + i] = c1[i]; }
  return r;
}

// RA = reuse-A hint: the 2nd..4th WMMA of a group sharing the same A fragment
// are emitted back-to-back (confirmed: 'matrix_a_reuse' in round-2 asm).
template <bool RA>
__device__ inline v8f wmma_bf16(v16bf a, v16bf b, v8f c) {
  return __builtin_amdgcn_wmma_f32_16x16x32_bf16(
      /*neg_a=*/false, a, /*neg_b=*/false, b,
      /*c_mod=*/(short)0, c, /*reuse_a=*/RA, /*reuse_b=*/false);
}

// 8 WMMAs for one k-step of a 32x64 wave tile, with reuse-A hints.
__device__ inline void mma_step(const v16bf (&af)[2], const v16bf (&bfr)[4],
                                v8f (&acc)[2][4]) {
#pragma unroll
  for (int mt = 0; mt < 2; mt++) {
    acc[mt][0] = wmma_bf16<false>(af[mt], bfr[0], acc[mt][0]);
    acc[mt][1] = wmma_bf16<true >(af[mt], bfr[1], acc[mt][1]);
    acc[mt][2] = wmma_bf16<true >(af[mt], bfr[2], acc[mt][2]);
    acc[mt][3] = wmma_bf16<true >(af[mt], bfr[3], acc[mt][3]);
  }
}

constexpr int PF = 256;  // prefetch distance in K elements (8 k-steps)

// WGP-scope prefetch via inline asm. __builtin_prefetch can only reach
// SE/SYS scope on this toolchain, and per the ISA only scope 0 (WGP) pulls
// the line into the WGP cache; SE/DEV/SYS stop at GL2 where our operands
// already live. Default scope field = 0 = WGP. No counter bookkeeping needed
// (prefetch does not touch LOADcnt).
__device__ inline void prefetch_wgp(const void* p) {
  asm volatile("global_prefetch_b8 %0, off" ::"v"(p));
}

// ---------------------------------------------------------------------------
// Precision conversion kernels
// ---------------------------------------------------------------------------

__global__ __launch_bounds__(256) void cast_f32_bf16(
    const float* __restrict__ in, bf16* __restrict__ out, int n4) {
  int i = blockIdx.x * 256 + threadIdx.x;
  if (i < n4) {
    v4f v = ((const v4f*)in)[i];
    v4bf o;
#pragma unroll
    for (int j = 0; j < 4; j++) o[j] = (bf16)v[j];
    ((v4bf*)out)[i] = o;
  }
}

// src [B,S,D] f32 -> srcT [B,D,S] bf16, LDS-tiled transpose (32x32 tiles)
__global__ __launch_bounds__(256) void transpose_to_bf16(
    const float* __restrict__ src, bf16* __restrict__ dstT) {
  __shared__ bf16 tile[32][34];  // 34 keeps column stride coprime with banks
  int b = blockIdx.z;
  int s0 = blockIdx.x * 32;
  int d0 = blockIdx.y * 32;
  const float* sp = src + (size_t)b * S * D;
  bf16* dp = dstT + (size_t)b * D * S;
  int x = threadIdx.x;  // 0..31
#pragma unroll
  for (int i = 0; i < 4; i++) {
    int y = threadIdx.y + i * 8;
    tile[y][x] = (bf16)sp[(size_t)(s0 + y) * D + d0 + x];
  }
  __syncthreads();
#pragma unroll
  for (int i = 0; i < 4; i++) {
    int y = threadIdx.y + i * 8;
    dp[(size_t)(d0 + y) * S + s0 + x] = tile[x][y];
  }
}

// ---------------------------------------------------------------------------
// GEMM1: logits[b,t,s] = sum_d tgt[b,t,d] * src[b,s,d]   (f32 out)
// A = tgt bf16 (M=S, K=D, lda=D); BT = src bf16 (N-major, ldbt=D)
// ---------------------------------------------------------------------------
__global__ __launch_bounds__(256) void gemm_logits(
    const bf16* __restrict__ tgtB, const bf16* __restrict__ srcB,
    float* __restrict__ logits) {
  const int bz = blockIdx.z;
  const bf16* A  = tgtB + (size_t)bz * S * D;
  const bf16* BT = srcB + (size_t)bz * S * D;
  float* C = logits + (size_t)bz * S * S;

  const int lane = threadIdx.x & 31;
  const int wave = threadIdx.x >> 5;
  const int m0 = blockIdx.y * 128 + (wave >> 1) * 32;
  const int n0 = blockIdx.x * 128 + (wave & 1) * 64;
  const int ml = lane & 15;
  const int kh = lane >> 4;

  v8f acc[2][4] = {};
  for (int k0 = 0; k0 < D; k0 += 32) {
    if (k0 + PF < D) {  // pull next cachelines into WGP$
      prefetch_wgp(A + (size_t)(m0 + ml) * D + k0 + PF);
      prefetch_wgp(BT + (size_t)(n0 + ml) * D + k0 + PF);
    }
    v16bf af[2], bfr[4];
#pragma unroll
    for (int mt = 0; mt < 2; mt++)
      af[mt] = load_a_bf16(A, D, m0 + mt * 16 + ml, kh, k0);
#pragma unroll
    for (int nt = 0; nt < 4; nt++)
      bfr[nt] = load_bt_bf16(BT, D, n0 + nt * 16 + ml, kh, k0);
    mma_step(af, bfr, acc);
  }
#pragma unroll
  for (int mt = 0; mt < 2; mt++)
#pragma unroll
    for (int nt = 0; nt < 4; nt++)
#pragma unroll
      for (int r = 0; r < 8; r++)
        C[(size_t)(m0 + mt * 16 + kh * 8 + r) * S + (n0 + nt * 16 + ml)] =
            acc[mt][nt][r];
}

// ---------------------------------------------------------------------------
// Softmax over axis=1 (the t axis): normalize each column s of [t,s] matrix.
// Online max+sum (1 read pass) + normalize (read+write pass): 3x64MB moved
// instead of 5x64MB. One thread per column; consecutive threads ->
// consecutive s (coalesced).
// ---------------------------------------------------------------------------
__global__ __launch_bounds__(256) void softmax_col(float* __restrict__ w) {
  const int b = blockIdx.y;
  const int s = blockIdx.x * 256 + threadIdx.x;
  float* p = w + (size_t)b * S * S + s;
  float m = -INFINITY, sum = 0.f;
  for (int t = 0; t < S; t++) {
    float x = p[(size_t)t * S];
    float mn = fmaxf(m, x);
    sum = sum * __expf(m - mn) + __expf(x - mn);
    m = mn;
  }
  float inv = 1.0f / sum;
  for (int t = 0; t < S; t++) {
    float x = p[(size_t)t * S];
    p[(size_t)t * S] = __expf(x - m) * inv;
  }
}

// ---------------------------------------------------------------------------
// GEMM2: wsum[b,t,d] = sum_s weight[b,t,s] * src[b,s,d]  (bf16 out)
// A = weight f32 (converted at fragment-load, lda=S); BT = srcT bf16 (ldbt=S)
// ---------------------------------------------------------------------------
__global__ __launch_bounds__(256) void gemm_wsum(
    const float* __restrict__ weight, const bf16* __restrict__ srcT,
    bf16* __restrict__ wsum) {
  const int bz = blockIdx.z;
  const float* A = weight + (size_t)bz * S * S;
  const bf16* BT = srcT + (size_t)bz * D * S;
  bf16* C = wsum + (size_t)bz * S * D;

  const int lane = threadIdx.x & 31;
  const int wave = threadIdx.x >> 5;
  const int m0 = blockIdx.y * 128 + (wave >> 1) * 32;
  const int n0 = blockIdx.x * 128 + (wave & 1) * 64;
  const int ml = lane & 15;
  const int kh = lane >> 4;

  v8f acc[2][4] = {};
  for (int k0 = 0; k0 < S; k0 += 32) {
    if (k0 + PF < S) {
      prefetch_wgp(A + (size_t)(m0 + ml) * S + k0 + PF);
      prefetch_wgp(BT + (size_t)(n0 + ml) * S + k0 + PF);
    }
    v16bf af[2], bfr[4];
#pragma unroll
    for (int mt = 0; mt < 2; mt++)
      af[mt] = load_a_f32(A, S, m0 + mt * 16 + ml, kh, k0);
#pragma unroll
    for (int nt = 0; nt < 4; nt++)
      bfr[nt] = load_bt_bf16(BT, S, n0 + nt * 16 + ml, kh, k0);
    mma_step(af, bfr, acc);
  }
#pragma unroll
  for (int mt = 0; mt < 2; mt++)
#pragma unroll
    for (int nt = 0; nt < 4; nt++)
#pragma unroll
      for (int r = 0; r < 8; r++)
        C[(size_t)(m0 + mt * 16 + kh * 8 + r) * D + (n0 + nt * 16 + ml)] =
            (bf16)acc[mt][nt][r];
}

// ---------------------------------------------------------------------------
// GEMM3: out[row,o] = sum_{k<2048} cat[row,k] * W[o,k] + b[o]
// cat(row, k) = k < D ? wsum[row,k] : tgt[row,k-D], rows flattened over B*S.
// BT = W bf16 row-major [O, 2D] (already N-major), ldbt = 2D.
// ---------------------------------------------------------------------------
__global__ __launch_bounds__(256) void gemm_out(
    const bf16* __restrict__ wsum, const bf16* __restrict__ tgtB,
    const bf16* __restrict__ wB, const float* __restrict__ bias,
    float* __restrict__ out) {
  const int lane = threadIdx.x & 31;
  const int wave = threadIdx.x >> 5;
  const int m0 = blockIdx.y * 128 + (wave >> 1) * 32;  // over B*S rows
  const int n0 = blockIdx.x * 128 + (wave & 1) * 64;   // over O cols
  const int ml = lane & 15;
  const int kh = lane >> 4;

  v8f acc[2][4] = {};
  for (int k0 = 0; k0 < 2 * D; k0 += 32) {
    const bf16* Ap = (k0 < D) ? wsum : tgtB;
    const int koff = (k0 < D) ? k0 : (k0 - D);
    if (koff + PF < D)
      prefetch_wgp(Ap + (size_t)(m0 + ml) * D + koff + PF);
    if (k0 + PF < 2 * D)
      prefetch_wgp(wB + (size_t)(n0 + ml) * 2 * D + k0 + PF);
    v16bf af[2], bfr[4];
#pragma unroll
    for (int mt = 0; mt < 2; mt++)
      af[mt] = load_a_bf16(Ap, D, m0 + mt * 16 + ml, kh, koff);
#pragma unroll
    for (int nt = 0; nt < 4; nt++)
      bfr[nt] = load_bt_bf16(wB, 2 * D, n0 + nt * 16 + ml, kh, k0);
    mma_step(af, bfr, acc);
  }
#pragma unroll
  for (int mt = 0; mt < 2; mt++)
#pragma unroll
    for (int nt = 0; nt < 4; nt++)
#pragma unroll
      for (int r = 0; r < 8; r++) {
        int row = m0 + mt * 16 + kh * 8 + r;
        int col = n0 + nt * 16 + ml;
        out[(size_t)row * O + col] = acc[mt][nt][r] + bias[col];
      }
}

// ---------------------------------------------------------------------------
extern "C" void kernel_launch(void* const* d_in, const int* in_sizes, int n_in,
                              void* d_out, int out_size, void* d_ws,
                              size_t ws_size, hipStream_t stream) {
  (void)in_sizes; (void)n_in; (void)out_size; (void)ws_size;
  const float* src  = (const float*)d_in[0];  // [B,S,D]
  const float* tgt  = (const float*)d_in[1];  // [B,S,D]
  const float* W    = (const float*)d_in[2];  // [O, 2D]
  const float* bias = (const float*)d_in[3];  // [O]

  float* out    = (float*)d_out;              // [B,S,O]
  float* weight = out + (size_t)Bn * S * O;   // [B,S,S] (post-softmax output)

  char* p = (char*)d_ws;
  bf16* srcB = (bf16*)p; p += (size_t)Bn * S * D * 2;   // 16 MB
  bf16* tgtB = (bf16*)p; p += (size_t)Bn * S * D * 2;   // 16 MB
  bf16* srcT = (bf16*)p; p += (size_t)Bn * S * D * 2;   // 16 MB
  bf16* wB   = (bf16*)p; p += (size_t)O * 2 * D * 2;    //  4 MB
  bf16* wsum = (bf16*)p; p += (size_t)Bn * S * D * 2;   // 16 MB

  const int nSD4 = Bn * S * D / 4;
  const int nW4  = O * 2 * D / 4;
  cast_f32_bf16<<<(nSD4 + 255) / 256, 256, 0, stream>>>(src, srcB, nSD4);
  cast_f32_bf16<<<(nSD4 + 255) / 256, 256, 0, stream>>>(tgt, tgtB, nSD4);
  cast_f32_bf16<<<(nW4 + 255) / 256, 256, 0, stream>>>(W, wB, nW4);
  transpose_to_bf16<<<dim3(S / 32, D / 32, Bn), dim3(32, 8), 0, stream>>>(
      src, srcT);

  gemm_logits<<<dim3(S / 128, S / 128, Bn), 256, 0, stream>>>(tgtB, srcB,
                                                              weight);
  softmax_col<<<dim3(S / 256, Bn), 256, 0, stream>>>(weight);
  gemm_wsum<<<dim3(D / 128, S / 128, Bn), 256, 0, stream>>>(weight, srcT,
                                                            wsum);
  gemm_out<<<dim3(O / 128, (Bn * S) / 128), 256, 0, stream>>>(wsum, tgtB, wB,
                                                              bias, out);
}
